// CMCI_Mamba_910533066819
// MI455X (gfx1250) — matrix operand order repaired and checked by reading, not compiled, not run on hardware
//
#include <hip/hip_runtime.h>
#include <hip/hip_bf16.h>
#include <cmath>

// CMCI-Mamba for MI455X (gfx1250, wave32).
// GEMMs: v_wmma_f32_16x16x32_f16, 32x(NT*16) output tile per wave
// (NT*2 WMMA per k-step), f16 operands loaded as b128 vectors.
// Selective scan fused into one kernel: 1024 chains x 16 states, lane = s.

#define DM 64
#define DI 128
#define DS 16
#define DR 4
#define KC 4
#define BSZ 8              // mamba batch = 2*B
#define SEQ 2048           // mamba sequence length (= C)
#define ROWS (BSZ * SEQ)   // 16384 GEMM rows
#define HALF_ELEMS (4 * 2048 * 64)  // one (B,C,L) tensor = 524288

typedef __attribute__((ext_vector_type(16))) _Float16 v16h;
typedef __attribute__((ext_vector_type(8)))  _Float16 v8h;
typedef __attribute__((ext_vector_type(8)))  float    v8f;

// ---------------------------------------------------------------------------
// GEMM: C[M x N] = A[M x K] * W[N x K]^T  (einsum 'mk,nk->mn'), f16 in, f32 out.
// One wave computes (MT*16) x (NT*16); A and B fragments each reused MT/NT ways.
// A 16x32 f16 layout: lane L -> row M=L&15, kh=L>>4,
//   elems 0..7 -> K=kh*8+e (one b128), elems 8..15 -> K=16+kh*8+(e-8) (one b128).
// B 32x16 f16 layout: lane L -> col N=L&15, same K order as A (two b128).
// C/D 16x16 f32: lane L -> col N=L&15, vgpr v -> row M=(L>>4)*8+v.
// GUARD: per-lane col<N masking (only the N=36 x-proj needs it).
// DUAL:  also emit an f16 copy of C (feeds the next layer's GEMM).
// ---------------------------------------------------------------------------
template <int NT, int MT, bool GUARD, bool DUAL>
__global__ __launch_bounds__(128)
void gemm_wmma_f16(const _Float16* __restrict__ A, const _Float16* __restrict__ W,
                   float* __restrict__ C, _Float16* __restrict__ C16,
                   int N, int Kd, int ldc, int ngrp)
{
    const int wave = (int)((blockIdx.x * blockDim.x + threadIdx.x) >> 5);
    const int lane = (int)(threadIdx.x & 31);
    const int tm  = wave / ngrp;
    const int tg  = wave - tm * ngrp;
    const int nb  = tg * NT * 16;             // first column of this wave's group
    const int ln  = lane & 15;
    const int kh  = lane >> 4;

    v8f acc[MT][NT];
#pragma unroll
    for (int mi = 0; mi < MT; ++mi)
#pragma unroll
        for (int t = 0; t < NT; ++t) acc[mi][t] = (v8f){};

    for (int k0 = 0; k0 < Kd; k0 += 32) {
        // ---- A fragments: two aligned 16B vector loads per M-tile ----
        v16h a[MT];
#pragma unroll
        for (int mi = 0; mi < MT; ++mi) {
            const int mr = (tm * MT + mi) * 16 + ln;
            const _Float16* Arow = A + (size_t)mr * Kd + k0;
            const v8h a0 = *(const v8h*)(Arow + kh * 8);
            const v8h a1 = *(const v8h*)(Arow + 16 + kh * 8);
#pragma unroll
            for (int e = 0; e < 8; ++e) { a[mi][e] = a0[e]; a[mi][8 + e] = a1[e]; }
        }
        // ---- B fragments, each reused across MT WMMAs ----
#pragma unroll
        for (int t = 0; t < NT; ++t) {
            const int nc = nb + t * 16 + ln;
            v16h b;
            if (!GUARD || nc < N) {
                const _Float16* Wrow = W + (size_t)nc * Kd + k0 + kh * 8;
                const v8h b0 = *(const v8h*)(Wrow);
                const v8h b1 = *(const v8h*)(Wrow + 16);
#pragma unroll
                for (int e = 0; e < 8; ++e) { b[e] = b0[e]; b[8 + e] = b1[e]; }
            } else {
#pragma unroll
                for (int e = 0; e < 16; ++e) b[e] = (_Float16)0.f;
            }
#pragma unroll
            for (int mi = 0; mi < MT; ++mi)
                acc[mi][t] = __builtin_amdgcn_wmma_f32_16x16x32_f16(
                    false, a[mi], false, b, (short)0, acc[mi][t], false, false);
        }
    }

#pragma unroll
    for (int mi = 0; mi < MT; ++mi) {
        const int mbase = (tm * MT + mi) * 16 + kh * 8;
#pragma unroll
        for (int t = 0; t < NT; ++t) {
            const int nc = nb + t * 16 + ln;
            if (!GUARD || nc < N) {
#pragma unroll
                for (int v = 0; v < 8; ++v) {
                    C[(size_t)(mbase + v) * ldc + nc] = acc[mi][t][v];
                    if (DUAL)
                        C16[(size_t)(mbase + v) * ldc + nc] =
                            (_Float16)acc[mi][t][v];
                }
            }
        }
    }
}

// ---------------------------------------------------------------------------
// Depthwise causal conv (K=4) over sequence dim + SiLU.
// Reads x = xz[:, 0:DI]; writes xconv f32 (for scan) and f16 (for x-proj GEMM).
// ---------------------------------------------------------------------------
__global__ __launch_bounds__(256)
void conv_silu(const float* __restrict__ xz, const float* __restrict__ convw,
               const float* __restrict__ convb,
               float* __restrict__ xcf, _Float16* __restrict__ xch)
{
    const int idx = (int)(blockIdx.x * blockDim.x + threadIdx.x); // ROWS*DI
    const int d = idx & (DI - 1);
    const int l = (idx >> 7) & (SEQ - 1);
    const int b = idx >> 18;
    float acc = convb[d];
#pragma unroll
    for (int k = 0; k < KC; ++k) {
        const int ls = l - (KC - 1) + k;
        const float xv = (ls >= 0)
            ? xz[((size_t)b * SEQ + ls) * (2 * DI) + d] : 0.f;
        acc += xv * convw[d * KC + k];
    }
    const float s = acc / (1.f + __expf(-acc));   // silu
    xcf[idx] = s;
    xch[idx] = (_Float16)s;
}

// ---------------------------------------------------------------------------
// Fused selective scan. lane&15 = state s, 2 chains (b,d) per wave.
// delta = softplus(dt@dt_w + dt_b); h = exp(delta*A)*h + delta*B*x;
// y = (sum_s h*C + D*x) * silu(z)  -> written f16 for the out-proj GEMM.
// ---------------------------------------------------------------------------
__global__ __launch_bounds__(256)
void mamba_scan(const float* __restrict__ xz, const float* __restrict__ xconv,
                const float* __restrict__ xdbl,
                const float* __restrict__ dtw, const float* __restrict__ dtb,
                const float* __restrict__ Alog, const float* __restrict__ Dp,
                _Float16* __restrict__ y)
{
    const int gtid  = (int)(blockIdx.x * blockDim.x + threadIdx.x);
    const int wave  = gtid >> 5;
    const int lane  = (int)(threadIdx.x & 31);
    const int s     = lane & 15;
    const int chain = wave * 2 + (lane >> 4);     // 0..1023
    const int b = chain >> 7;
    const int d = chain & (DI - 1);

    const float Asd = -__expf(Alog[d * DS + s]);
    const float w0 = dtw[d * DR + 0], w1 = dtw[d * DR + 1];
    const float w2 = dtw[d * DR + 2], w3 = dtw[d * DR + 3];
    const float bias = dtb[d];
    const float Dd   = Dp[d];

    float h = 0.f;
    const size_t base = (size_t)b * SEQ;
    for (int l = 0; l < SEQ; ++l) {
        const size_t r = base + l;
        const float* xr = xdbl + r * 48;          // padded row stride
        const float dpre = xr[0]*w0 + xr[1]*w1 + xr[2]*w2 + xr[3]*w3 + bias;
        const float delta = (dpre > 20.f) ? dpre
                          : __logf(1.f + __expf(dpre));   // softplus
        const float Bm = xr[DR + s];
        const float Cm = xr[DR + DS + s];
        const float xv = xconv[r * DI + d];
        h = __expf(delta * Asd) * h + delta * Bm * xv;
        float contrib = h * Cm;
#pragma unroll
        for (int off = 1; off < 16; off <<= 1)    // reduce over 16 states
            contrib += __shfl_xor(contrib, off, 32);
        if (s == 0) {
            const float z = xz[r * (2 * DI) + DI + d];
            y[r * DI + d] =
                (_Float16)((contrib + Dd * xv) * (z / (1.f + __expf(-z))));
        }
    }
}

// ---------------------------------------------------------------------------
// f32 -> f16 conversion (weights, once per layer; tiny)
// ---------------------------------------------------------------------------
__global__ __launch_bounds__(256)
void convert_h(const float* __restrict__ in, _Float16* __restrict__ out, int n)
{
    const int idx = (int)(blockIdx.x * blockDim.x + threadIdx.x);
    if (idx < n) out[idx] = (_Float16)in[idx];
}

// cf1/cf2 builder (first branch): cf1=[Ms[:h];Pan[h:]], cf2=[Pan[:h];Ms[h:]] (f16)
__global__ __launch_bounds__(256)
void build_u_first(const float* __restrict__ Ms, const float* __restrict__ Pan,
                   _Float16* __restrict__ u)
{
    const int idx = (int)(blockIdx.x * blockDim.x + threadIdx.x); // HALF_ELEMS
    const int c = (idx >> 6) & (SEQ - 1);
    const bool lo = c < (SEQ / 2);
    u[idx]              = (_Float16)(lo ? Ms[idx]  : Pan[idx]);
    u[idx + HALF_ELEMS] = (_Float16)(lo ? Pan[idx] : Ms[idx]);
}

// cf3/cf4 builder (second branch): interleave even/odd rows of Pan and Ms (f16)
__global__ __launch_bounds__(256)
void build_u_second(const float* __restrict__ Ms, const float* __restrict__ Pan,
                    _Float16* __restrict__ u)
{
    const int idx = (int)(blockIdx.x * blockDim.x + threadIdx.x);
    const int c = (idx >> 6) & (SEQ - 1);
    const bool even = (c & 1) == 0;
    u[idx]              = (_Float16)(even ? Pan[idx] : Ms[idx]);   // cf3
    u[idx + HALF_ELEMS] = (_Float16)(even ? Ms[idx]  : Pan[idx]);  // cf4
}

// tgt = relu((out[:4] + out[4:]) * 0.5 + tgt)
__global__ __launch_bounds__(256)
void post_layer(const float* __restrict__ out, float* __restrict__ tgt)
{
    const int idx = (int)(blockIdx.x * blockDim.x + threadIdx.x);
    const float v = 0.5f * (out[idx] + out[idx + HALF_ELEMS]) + tgt[idx];
    tgt[idx] = v > 0.f ? v : 0.f;
}

// ---------------------------------------------------------------------------
extern "C" void kernel_launch(void* const* d_in, const int* in_sizes, int n_in,
                              void* d_out, int out_size, void* d_ws, size_t ws_size,
                              hipStream_t stream)
{
    (void)in_sizes; (void)n_in; (void)out_size; (void)ws_size;

    float* Ms  = (float*)d_out;            // updated in place, also final output
    float* Pan = Ms + HALF_ELEMS;
    hipMemcpyAsync(Ms,  d_in[0], HALF_ELEMS * sizeof(float),
                   hipMemcpyDeviceToDevice, stream);
    hipMemcpyAsync(Pan, d_in[1], HALF_ELEMS * sizeof(float),
                   hipMemcpyDeviceToDevice, stream);

    float* ws = (float*)d_ws;
    // f32 buffers
    float* outF = ws;                                  // (8,2048,64)  1,048,576
    float* xz   = outF + 2 * HALF_ELEMS;               // (16384,256)  4,194,304
    float* xcf  = xz   + (size_t)ROWS * 2 * DI;        // (16384,128)  2,097,152
    float* xd   = xcf  + (size_t)ROWS * DI;            // (16384,48)     786,432
    // f16 buffers (counted in float slots: halfs/2)
    _Float16* u16a = (_Float16*)(xd + (size_t)ROWS * 48);          // (16384,64)
    _Float16* u16b = u16a + (size_t)ROWS * DM;                     // (16384,64)
    _Float16* xch  = u16b + (size_t)ROWS * DM;                     // (16384,128)
    _Float16* ybh  = xch  + (size_t)ROWS * DI;                     // (16384,128)
    _Float16* w16  = ybh  + (size_t)ROWS * DI;                     // <= 16384 halfs

    auto run_mamba = [&](const _Float16* u, float* out, _Float16* out16,
                         int pi, int layer) {
        const float* in_w  = (const float*)d_in[pi + 0] + (size_t)layer * 2 * DI * DM;
        const float* convw = (const float*)d_in[pi + 1] + (size_t)layer * DI * KC;
        const float* convb = (const float*)d_in[pi + 2] + (size_t)layer * DI;
        const float* xp_w  = (const float*)d_in[pi + 3] + (size_t)layer * (DR + 2 * DS) * DI;
        const float* dt_w  = (const float*)d_in[pi + 4] + (size_t)layer * DI * DR;
        const float* dt_b  = (const float*)d_in[pi + 5] + (size_t)layer * DI;
        const float* Alog  = (const float*)d_in[pi + 6] + (size_t)layer * DI * DS;
        const float* Dp    = (const float*)d_in[pi + 7] + (size_t)layer * DI;
        const float* out_w = (const float*)d_in[pi + 8] + (size_t)layer * DM * DI;

        // xz = u @ in_w^T : (16384x64)x(256x64)^T ; 512 M-tiles(32) x 4 groups(64)
        convert_h<<<64, 256, 0, stream>>>(in_w, w16, 2 * DI * DM);
        gemm_wmma_f16<4, 2, false, false><<<512, 128, 0, stream>>>(
            u, w16, xz, (_Float16*)nullptr, 2 * DI, DM, 2 * DI, 4);

        // depthwise conv + silu (f32 + f16 outputs)
        conv_silu<<<(ROWS * DI) / 256, 256, 0, stream>>>(xz, convw, convb, xcf, xch);

        // xdbl = x @ xp_w^T : (16384x128)x(36x128)^T ; NT=3 with N-tail guard
        convert_h<<<18, 256, 0, stream>>>(xp_w, w16, (DR + 2 * DS) * DI);
        gemm_wmma_f16<3, 2, true, false><<<128, 128, 0, stream>>>(
            xch, w16, xd, (_Float16*)nullptr, DR + 2 * DS, DI, 48, 1);

        // fused selective scan (+softplus, +gate) -> f16 y
        mamba_scan<<<64, 256, 0, stream>>>(xz, xcf, xd, dt_w, dt_b, Alog, Dp, ybh);

        // out = y @ out_w^T : (16384x128)x(64x128)^T ; writes f32 + f16(next u)
        convert_h<<<32, 256, 0, stream>>>(out_w, w16, DM * DI);
        gemm_wmma_f16<4, 2, false, true><<<128, 128, 0, stream>>>(
            ybh, w16, out, out16, DM, DI, DM, 1);
    };

    const int EB = HALF_ELEMS / 256;  // 2048 blocks for elementwise kernels

    // ---- branch A: update Ms (params a_* at d_in[2..10]) ----
    build_u_first<<<EB, 256, 0, stream>>>(Ms, Pan, u16a);
    run_mamba(u16a, outF, u16b, 2, 0);
    post_layer<<<EB, 256, 0, stream>>>(outF, Ms);
    run_mamba(u16b, outF, u16a, 2, 1);        // next-layer input = prev output
    post_layer<<<EB, 256, 0, stream>>>(outF, Ms);

    // ---- branch B: update Pan (params b_* at d_in[11..19]) ----
    build_u_second<<<EB, 256, 0, stream>>>(Ms, Pan, u16a);
    run_mamba(u16a, outF, u16b, 11, 0);
    post_layer<<<EB, 256, 0, stream>>>(outF, Pan);
    run_mamba(u16b, outF, u16a, 11, 1);
    post_layer<<<EB, 256, 0, stream>>>(outF, Pan);
}
